// PatchEmbedding_7894149890251
// MI455X (gfx1250) — matrix operand order repaired and checked by reading, not compiled, not run on hardware
//
#include <hip/hip_runtime.h>

typedef __attribute__((ext_vector_type(16))) _Float16 v16h;
typedef __attribute__((ext_vector_type(2)))  _Float16 h2;
typedef __attribute__((ext_vector_type(8)))  float    v8f;
typedef int v4i __attribute__((ext_vector_type(4)));

constexpr int BQ     = 8;
constexpr int NPTS   = 131072;          // 2^17
constexpr int PATCHN = 32;
constexpr int STRD   = 16;
constexpr int LWIN   = 8191;
constexpr int NTOK   = BQ * LWIN;       // 65528
constexpr int NTOKP  = 65536;           // padded rows for GEMM tiles
constexpr int DM     = 256;

// ---- workspace layout (bytes) ----
constexpr size_t OFF_KEYS   = 0;                                   // u64 keys  8 MB
constexpr size_t OFF_SP     = OFF_KEYS + (size_t)BQ*NPTS*8;        // sorted pts 12.6 MB
constexpr size_t OFF_MM     = OFF_SP   + (size_t)BQ*NPTS*3*4;      // min/max 192B
constexpr size_t OFF_COMB   = OFF_MM   + 256;                      // combined f16 32 MB
constexpr size_t OFF_W2T    = OFF_COMB + (size_t)NTOKP*DM*2;
constexpr size_t OFF_WP2T   = OFF_W2T  + 64*128*2;
constexpr size_t OFF_WPROJT = OFF_WP2T + 64*128*2;

// ---- typed address-space pointers for async global->LDS DMA ----
typedef v4i __attribute__((address_space(1)))* gv4i_p;   // global v4i*
typedef v4i __attribute__((address_space(3)))* lv4i_p;   // LDS v4i*

#if __has_builtin(__builtin_amdgcn_global_load_async_to_lds_b128)
#define HAVE_ASYNC_LDS 1
#else
#define HAVE_ASYNC_LDS 0
#endif

__device__ __forceinline__ void wait_asynccnt0() {
#if __has_builtin(__builtin_amdgcn_s_wait_asynccnt)
  __builtin_amdgcn_s_wait_asynccnt(0);
#else
  asm volatile("s_wait_asynccnt 0" ::: "memory");
#endif
}

// Cooperative async preload of a [128 rows x 64 halfs] f16 matrix (row-major,
// global ld=64) into LDS with padded row stride 68. 16B granules: 8 per row,
// 1024 total, 4 per thread at 256 threads. Tracked by ASYNCcnt.
__device__ __forceinline__ void preload_w_tile(const _Float16* __restrict__ g,
                                               _Float16* s /*LDS*/, int t) {
#if HAVE_ASYNC_LDS
  for (int i = t; i < 1024; i += 256) {
    int n = i >> 3, c = i & 7;
    __builtin_amdgcn_global_load_async_to_lds_b128(
        (gv4i_p)(void*)(g + n * 64 + c * 8),
        (lv4i_p)(void*)(s + n * 68 + c * 8), 0, 0);
  }
#else
  for (int i = t; i < 128 * 64; i += 256) {
    int n = i >> 6, k = i & 63;
    s[n * 68 + k] = g[i];
  }
#endif
}

// ---------------- WMMA fragment loaders (contiguous k = 16*(lane/16) + s, same in A and B) ----------------
// A: 16x32 f16, rows M0..M0+15, cols K0..K0+31, from row-major [*, ld]
__device__ __forceinline__ v16h ldA(const _Float16* base, int ld, int M0, int K0, int lane) {
  int m  = M0 + (lane & 15);
  int kb = K0 + ((lane >> 4) << 4);          // lanes 0-15: K 0..15, lanes 16-31: K 16..31
  const _Float16* row = base + (size_t)m * ld;
  v16h a;
#pragma unroll
  for (int r = 0; r < 8; ++r) {
    int k = kb + 2 * r;
    h2 p = *(const h2*)(row + k);
    a[2 * r] = p.x; a[2 * r + 1] = p.y;
  }
  return a;
}
// B: 32x16 f16, rows K0..K0+31, cols N0..N0+15, from TRANSPOSED [N][K] row-major (ld = K-stride)
__device__ __forceinline__ v16h ldBT(const _Float16* baseT, int ld, int K0, int N0, int lane) {
  int n  = N0 + (lane & 15);
  int kh = K0 + ((lane >> 4) << 4);          // lanes 0-15: K 0..15, lanes 16-31: K 16..31
  const _Float16* row = baseT + (size_t)n * ld;
  v16h b;
#pragma unroll
  for (int r = 0; r < 8; ++r) {
    h2 p = *(const h2*)(row + kh + 2 * r);
    b[2 * r] = p.x; b[2 * r + 1] = p.y;
  }
  return b;
}
#define WMMA_F32_F16(A, Bf, C) \
  __builtin_amdgcn_wmma_f32_16x16x32_f16(false, (A), false, (Bf), (short)0, (C), false, false)

// ---------------- weight convert + transpose: dst[n*K+k] = (f16)src[k*N+n] ----
__global__ void k_cvt_t(const float* __restrict__ src, _Float16* __restrict__ dst, int K, int N) {
  int gid = blockIdx.x * 256 + threadIdx.x;
  if (gid >= K * N) return;
  int k = gid / N, n = gid % N;
  dst[(size_t)n * K + k] = (_Float16)src[gid];
}

// ---------------- per-batch coordinate min/max ----------------
__global__ void k_minmax(const float* __restrict__ pts, float* __restrict__ mm) {
  __shared__ float s[6][256];
  int b = blockIdx.x, t = threadIdx.x;
  const float* p = pts + (size_t)b * NPTS * 3;
  float mn[3] = {1e30f, 1e30f, 1e30f}, mx[3] = {-1e30f, -1e30f, -1e30f};
  for (int i = t; i < NPTS; i += 256) {
#pragma unroll
    for (int c = 0; c < 3; ++c) {
      float v = p[3 * i + c];
      mn[c] = fminf(mn[c], v); mx[c] = fmaxf(mx[c], v);
    }
  }
#pragma unroll
  for (int c = 0; c < 3; ++c) { s[c][t] = mn[c]; s[3 + c][t] = mx[c]; }
  __syncthreads();
  for (int off = 128; off > 0; off >>= 1) {
    if (t < off) {
#pragma unroll
      for (int c = 0; c < 3; ++c) {
        s[c][t]     = fminf(s[c][t],     s[c][t + off]);
        s[3 + c][t] = fmaxf(s[3 + c][t], s[3 + c][t + off]);
      }
    }
    __syncthreads();
  }
  if (t == 0) {
#pragma unroll
    for (int c = 0; c < 3; ++c) { mm[b * 6 + c] = s[c][0]; mm[b * 6 + 3 + c] = s[3 + c][0]; }
  }
}

// ---------------- Morton key (30 bits) << 17 | point index (17 bits) ----------
__device__ __forceinline__ unsigned long long expand_bits(unsigned long long v) {
  v = (v | v << 32) & 8725724278095871ULL;
  v = (v | v << 16) & 8725728556220671ULL;
  v = (v | v << 8)  & 1157144660301377551ULL;
  v = (v | v << 4)  & 1207822528635744451ULL;
  v = (v | v << 2)  & 1317624576693539401ULL;
  return v;
}
__global__ void k_morton(const float* __restrict__ pts, const float* __restrict__ mm,
                         unsigned long long* __restrict__ keys) {
  int gid = blockIdx.x * 256 + threadIdx.x;        // exactly B*NPTS threads
  int b = gid >> 17, i = gid & (NPTS - 1);
  const float* p = pts + (size_t)gid * 3;
  unsigned long long q[3];
#pragma unroll
  for (int c = 0; c < 3; ++c) {
    float mn = mm[b * 6 + c], mx = mm[b * 6 + 3 + c];
    float sc = fmaxf(mx - mn, 1e-8f);
    long long qi = (long long)(((p[c] - mn) / sc) * 1023.0f);   // trunc, like astype(int64)
    qi = qi < 0 ? 0 : (qi > 1023 ? 1023 : qi);
    q[c] = (unsigned long long)qi;
  }
  unsigned long long key = (expand_bits(q[0]) << 2) | (expand_bits(q[1]) << 1) | expand_bits(q[2]);
  keys[gid] = (key << 17) | (unsigned long long)i;
}

// ---------------- bitonic sort (u64 keys, per-batch direction) ----------------
// local phase: fully sort 2048-chunks through stages k = 2 .. 2048 in LDS
__global__ void k_bitonic_local(unsigned long long* __restrict__ keys) {
  __shared__ unsigned long long s[2048];
  unsigned t = threadIdx.x;                        // 1024 threads
  size_t base = (size_t)blockIdx.x * 2048;
  s[t] = keys[base + t]; s[t + 1024] = keys[base + t + 1024];
  for (unsigned k = 2; k <= 2048; k <<= 1) {
    for (unsigned j = k >> 1; j >= 1; j >>= 1) {
      __syncthreads();
      unsigned i = ((t & ~(j - 1)) << 1) | (t & (j - 1));
      unsigned il = (unsigned)((base + i) & (NPTS - 1));
      bool asc = (il & k) == 0;
      unsigned long long x = s[i], y = s[i + j];
      if ((x > y) == asc) { s[i] = y; s[i + j] = x; }
    }
  }
  __syncthreads();
  keys[base + t] = s[t]; keys[base + t + 1024] = s[t + 1024];
}
// one global compare-exchange pass (j >= 2048)
__global__ void k_bitonic_global(unsigned long long* __restrict__ keys, unsigned j, unsigned k) {
  unsigned t = blockIdx.x * 256 + threadIdx.x;     // exactly B*NPTS/2 threads
  unsigned i = ((t & ~(j - 1)) << 1) | (t & (j - 1));
  unsigned long long x = keys[i], y = keys[i + j];
  bool asc = ((i & (NPTS - 1)) & k) == 0;
  if ((x > y) == asc) { keys[i] = y; keys[i + j] = x; }
}
// finish a stage: j = 1024 .. 1 in LDS for outer stage k
__global__ void k_bitonic_finish(unsigned long long* __restrict__ keys, unsigned k) {
  __shared__ unsigned long long s[2048];
  unsigned t = threadIdx.x;
  size_t base = (size_t)blockIdx.x * 2048;
  s[t] = keys[base + t]; s[t + 1024] = keys[base + t + 1024];
  bool asc = (((unsigned)(base & (NPTS - 1))) & k) == 0;       // k > 2048: uniform per chunk
  for (unsigned j = 1024; j >= 1; j >>= 1) {
    __syncthreads();
    unsigned i = ((t & ~(j - 1)) << 1) | (t & (j - 1));
    unsigned long long x = s[i], y = s[i + j];
    if ((x > y) == asc) { s[i] = y; s[i + j] = x; }
  }
  __syncthreads();
  keys[base + t] = s[t]; keys[base + t + 1024] = s[t + 1024];
}

// ---------------- gather sorted points + emit sort_idx (int64) ----------------
__global__ void k_gather(const float* __restrict__ pts, const unsigned long long* __restrict__ keys,
                         float* __restrict__ sp, long long* __restrict__ sidx) {
  int gid = blockIdx.x * 256 + threadIdx.x;
  int b = gid >> 17;
  unsigned sidx_i = (unsigned)(keys[gid] & 0x1FFFFULL);
  const float* src = pts + ((size_t)b * NPTS + sidx_i) * 3;
  float* dst = sp + (size_t)gid * 3;
  dst[0] = src[0]; dst[1] = src[1]; dst[2] = src[2];
  sidx[gid] = (long long)sidx_i;
}

// ---------------- geo path: centroid, 3->64 (VALU) -> 64->128 (WMMA) -> maxpool
__global__ void __launch_bounds__(256) k_geo(
    const float* __restrict__ sp, const float* __restrict__ W1, const float* __restrict__ b1,
    const _Float16* __restrict__ W2T, const float* __restrict__ b2,
    _Float16* __restrict__ comb, float* __restrict__ cent) {
  __shared__ _Float16 sh[8 * 32 * 68];             // 8 windows x (32 x 64, row-pad 68)
  __shared__ _Float16 sW2t[128 * 68];              // W2 transposed [N=128][K=64], pad 68
  __shared__ float sW1[192], sb1[64], sb2[128];
  int t = threadIdx.x, wave = t >> 5, lane = t & 31;
  preload_w_tile(W2T, sW2t, t);                    // async DMA global->LDS (ASYNCcnt)
  if (t < 192) sW1[t] = W1[t];
  if (t < 64)  sb1[t] = b1[t];
  if (t < 128) sb2[t] = b2[t];

  // overlap DMA with the first-layer work (only sh/sW1/sb1 consumed here)
  int w = blockIdx.x * 8 + wave;                   // < NTOK exactly (8191*8)
  int b = w / LWIN, l = w % LWIN;
  const float* p = sp + ((size_t)b * NPTS + (size_t)l * STRD + lane) * 3;
  float px = p[0], py = p[1], pz = p[2];
  float sx = px, sy = py, sz = pz;
#pragma unroll
  for (int m = 16; m >= 1; m >>= 1) {
    sx += __shfl_xor(sx, m); sy += __shfl_xor(sy, m); sz += __shfl_xor(sz, m);
  }
  float cx = sx * (1.0f / 32.0f), cy = sy * (1.0f / 32.0f), cz = sz * (1.0f / 32.0f);
  if (lane == 0) { cent[(size_t)w * 3] = cx; cent[(size_t)w * 3 + 1] = cy; cent[(size_t)w * 3 + 2] = cz; }
  float lx = px - cx, ly = py - cy, lz = pz - cz;
  __syncthreads();                                 // sW1/sb1 visible
  _Float16* hrow = sh + wave * (32 * 68) + lane * 68;
#pragma unroll 8
  for (int d = 0; d < 64; ++d) {
    float v = sb1[d] + lx * sW1[d] + ly * sW1[64 + d] + lz * sW1[128 + d];
    hrow[d] = (_Float16)fmaxf(v, 0.0f);
  }
#if HAVE_ASYNC_LDS
  wait_asynccnt0();                                // my async W2 granules landed
#endif
  __syncthreads();                                 // everyone's h rows + W2 tile visible

  const _Float16* hbase = sh + wave * (32 * 68);
  for (int nt = 0; nt < 8; ++nt) {
    v8f acc0 = {}, acc1 = {};
#pragma unroll
    for (int ks = 0; ks < 2; ++ks) {
      v16h bb = ldBT(sW2t, 68, ks * 32, nt * 16, lane);
      v16h a0 = ldA(hbase, 68, 0,  ks * 32, lane);
      v16h a1 = ldA(hbase, 68, 16, ks * 32, lane);
      acc0 = WMMA_F32_F16(a0, bb, acc0);
      acc1 = WMMA_F32_F16(a1, bb, acc1);
    }
    float bias = sb2[nt * 16 + (lane & 15)];
    float cm = -1e30f;
#pragma unroll
    for (int r = 0; r < 8; ++r) {
      cm = fmaxf(cm, fmaxf(acc0[r] + bias, 0.0f));
      cm = fmaxf(cm, fmaxf(acc1[r] + bias, 0.0f));
    }
    cm = fmaxf(cm, __shfl_xor(cm, 16));            // merge row halves (same column)
    if (lane < 16) comb[(size_t)w * DM + nt * 16 + lane] = (_Float16)cm;
  }
}

// ---------------- pos path: relu(cen@Wp1+bp1) (VALU) @ Wp2 + bp2 (WMMA) -------
__global__ void __launch_bounds__(256) k_pos(
    const float* __restrict__ cent, const float* __restrict__ Wp1, const float* __restrict__ bp1,
    const _Float16* __restrict__ Wp2T, const float* __restrict__ bp2, _Float16* __restrict__ comb) {
  __shared__ _Float16 sh1[128 * 68];               // 128 tokens x 64, pad 68
  __shared__ _Float16 sW[128 * 68];
  __shared__ float sWp1[192], sbp1[64], sbp2[128];
  int t = threadIdx.x;
  preload_w_tile(Wp2T, sW, t);                     // async DMA global->LDS (ASYNCcnt)
  if (t < 192) sWp1[t] = Wp1[t];
  if (t < 64)  sbp1[t] = bp1[t];
  if (t < 128) sbp2[t] = bp2[t];
  __syncthreads();

  int tok = t >> 1, half = t & 1;
  int w = blockIdx.x * 128 + tok;
  float cx = 0.f, cy = 0.f, cz = 0.f;
  if (w < NTOK) { cx = cent[(size_t)w * 3]; cy = cent[(size_t)w * 3 + 1]; cz = cent[(size_t)w * 3 + 2]; }
#pragma unroll 8
  for (int dd = 0; dd < 32; ++dd) {
    int d = half * 32 + dd;
    float v = sbp1[d] + cx * sWp1[d] + cy * sWp1[64 + d] + cz * sWp1[128 + d];
    sh1[tok * 68 + d] = (_Float16)fmaxf(v, 0.0f);
  }
#if HAVE_ASYNC_LDS
  wait_asynccnt0();
#endif
  __syncthreads();

  int wave = t >> 5, lane = t & 31;
  const _Float16* hbase = sh1 + wave * (16 * 68);
  v16h a0 = ldA(hbase, 68, 0, 0, lane);
  v16h a1 = ldA(hbase, 68, 0, 32, lane);
  for (int nt = 0; nt < 8; ++nt) {
    v8f acc = {};
    acc = WMMA_F32_F16(a0, ldBT(sW, 68, 0,  nt * 16, lane), acc);
    acc = WMMA_F32_F16(a1, ldBT(sW, 68, 32, nt * 16, lane), acc);
    int col = nt * 16 + (lane & 15);
    float bias = sbp2[col];
#pragma unroll
    for (int r = 0; r < 8; ++r) {
      int row = wave * 16 + r + ((lane >> 4) << 3);
      int ww = blockIdx.x * 128 + row;
      if (ww < NTOK) comb[(size_t)ww * DM + 128 + col] = (_Float16)(acc[r] + bias);
    }
  }
}

// ---------------- projection GEMM: tokens = combined @ Wproj + bproj ----------
__global__ void __launch_bounds__(256) k_proj(
    const _Float16* __restrict__ comb, const _Float16* __restrict__ WpT,
    const float* __restrict__ bproj, float* __restrict__ out) {
  int wave = threadIdx.x >> 5, lane = threadIdx.x & 31;
  int mt = blockIdx.x * 8 + wave;                  // 4096 M-tiles
  __builtin_prefetch(WpT, 0, 3);
  const _Float16* abase = comb + (size_t)mt * 16 * DM;
  v16h a[8];
#pragma unroll
  for (int ks = 0; ks < 8; ++ks) a[ks] = ldA(abase, DM, 0, ks * 32, lane);
  for (int nt = 0; nt < 16; ++nt) {
    v8f acc = {};
#pragma unroll
    for (int ks = 0; ks < 8; ++ks) {
      v16h bb = ldBT(WpT, DM, ks * 32, nt * 16, lane);
      acc = WMMA_F32_F16(a[ks], bb, acc);
    }
    int col = nt * 16 + (lane & 15);
    float bias = bproj[col];
#pragma unroll
    for (int r = 0; r < 8; ++r) {
      int row = mt * 16 + r + ((lane >> 4) << 3);
      if (row < NTOK) out[(size_t)row * DM + col] = acc[r] + bias;
    }
  }
}

// ------------------------------- launcher -------------------------------------
extern "C" void kernel_launch(void* const* d_in, const int* in_sizes, int n_in,
                              void* d_out, int out_size, void* d_ws, size_t ws_size,
                              hipStream_t stream) {
  const float* points = (const float*)d_in[0];
  const float* W1    = (const float*)d_in[1];
  const float* b1    = (const float*)d_in[2];
  const float* W2    = (const float*)d_in[3];
  const float* b2    = (const float*)d_in[4];
  const float* Wp1   = (const float*)d_in[5];
  const float* bp1   = (const float*)d_in[6];
  const float* Wp2   = (const float*)d_in[7];
  const float* bp2   = (const float*)d_in[8];
  const float* Wproj = (const float*)d_in[9];
  const float* bproj = (const float*)d_in[10];

  char* ws = (char*)d_ws;
  unsigned long long* keys = (unsigned long long*)(ws + OFF_KEYS);
  float*     sp    = (float*)(ws + OFF_SP);
  float*     mm    = (float*)(ws + OFF_MM);
  _Float16*  comb  = (_Float16*)(ws + OFF_COMB);
  _Float16*  W2T   = (_Float16*)(ws + OFF_W2T);
  _Float16*  Wp2T  = (_Float16*)(ws + OFF_WP2T);
  _Float16*  WprT  = (_Float16*)(ws + OFF_WPROJT);

  float* tokens = (float*)d_out;
  float* cent   = tokens + (size_t)NTOK * DM;
  long long* sidx = (long long*)(cent + (size_t)NTOK * 3);

  // weight convert/transpose (independent of sort)
  k_cvt_t<<<32,  256, 0, stream>>>(W2,    W2T,  64, 128);
  k_cvt_t<<<32,  256, 0, stream>>>(Wp2,   Wp2T, 64, 128);
  k_cvt_t<<<256, 256, 0, stream>>>(Wproj, WprT, 256, 256);

  // morton + sort
  k_minmax<<<BQ, 256, 0, stream>>>(points, mm);
  k_morton<<<(BQ * NPTS) / 256, 256, 0, stream>>>(points, mm, keys);
  k_bitonic_local<<<(BQ * NPTS) / 2048, 1024, 0, stream>>>(keys);
  for (unsigned k = 4096; k <= (unsigned)NPTS; k <<= 1) {
    for (unsigned j = k >> 1; j >= 2048; j >>= 1)
      k_bitonic_global<<<(BQ * NPTS / 2) / 256, 256, 0, stream>>>(keys, j, k);
    k_bitonic_finish<<<(BQ * NPTS) / 2048, 1024, 0, stream>>>(keys, k);
  }
  k_gather<<<(BQ * NPTS) / 256, 256, 0, stream>>>(points, keys, sp, sidx);

  // MLPs + projection
  k_geo <<<NTOK / 8,    256, 0, stream>>>(sp, W1, b1, W2T, b2, comb, cent);
  k_pos <<<NTOKP / 128, 256, 0, stream>>>(cent, Wp1, bp1, Wp2T, bp2, comb);
  k_proj<<<(NTOKP / 16) / 8, 256, 0, stream>>>(comb, WprT, bproj, tokens);
}
